// DeleteEdgeDecoder_51445118271770
// MI455X (gfx1250) — matrix operand order repaired and checked by reading, not compiled, not run on hardware
//
#include <hip/hip_runtime.h>
#include <hip/hip_bf16.h>

typedef _Float16 h16v __attribute__((ext_vector_type(16)));
typedef _Float16 h8v  __attribute__((ext_vector_type(8)));
typedef _Float16 h4v  __attribute__((ext_vector_type(4)));
typedef float    f8v  __attribute__((ext_vector_type(8)));
typedef float    f4v  __attribute__((ext_vector_type(4)));

#define BB      128
#define NN      2000
#define EE      2000
#define EMBED   256
#define KDIM    512      // 2*EMBED (f16 GEMM part of the 513-dim features)
#define HIDDEN  512
#define MTILE   64       // edges per block (4 groups of 16)
#define LDS_STRIDE 520   // 16-byte-aligned row stride in halves, de-banked

// ---- prep: W1[k][n] (f32, k<512) -> W1t[n][k] (f16, n-major) ----------------
__global__ void w1_transpose_f16(const float* __restrict__ W1,
                                 _Float16* __restrict__ W1t) {
    int idx = blockIdx.x * blockDim.x + threadIdx.x;     // 0 .. 512*512-1
    if (idx >= KDIM * HIDDEN) return;
    int n = idx >> 9;
    int k = idx & 511;
    W1t[n * KDIM + k] = (_Float16)W1[k * HIDDEN + n];
}

// ---- main: 64 edges per block, 8 waves x 64 hidden cols ---------------------
__global__ __launch_bounds__(256)
void edge_mlp_wmma(const float* __restrict__ emb,      // (B,N,256) f32
                   const float* __restrict__ locs,     // (B,N,2)   f32
                   const long long* __restrict__ el,   // (B,E,2)   i64
                   const int* __restrict__ dbias,      // scalar
                   const float* __restrict__ W1,       // (513,512) f32 (row 512 used)
                   const float* __restrict__ b1,       // (512)
                   const float* __restrict__ W2,       // (512)
                   const float* __restrict__ b2,       // (1)
                   const _Float16* __restrict__ W1t,   // (512,512) f16, n-major
                   float* __restrict__ out)            // (B,E) f32
{
    __shared__ __align__(16) _Float16 ldsA[MTILE * LDS_STRIDE];   // ~65 KB
    __shared__ int   nodeI[MTILE], nodeJ[MTILE], validm[MTILE];
    __shared__ float distm[MTILE];
    __shared__ float slog[MTILE];

    const int tid = threadIdx.x;
    const int b   = blockIdx.y;
    const int e0  = blockIdx.x * MTILE;

    // stage 1: indices, validity, distance, logit init (guarded: E % 64 != 0)
    if (tid < MTILE) {
        const int m = tid;
        const int e = e0 + m;
        long long i = -1, j = -1;
        if (e < EE) {
            i = el[(((long long)b * EE) + e) * 2 + 0];
            j = el[(((long long)b * EE) + e) * 2 + 1];
        }
        validm[m] = (i >= 0) && (j >= 0);
        int ic = i < 0 ? 0 : (int)i;
        int jc = j < 0 ? 0 : (int)j;
        nodeI[m] = ic;  nodeJ[m] = jc;
        float dx = locs[((long long)b * NN + ic) * 2 + 0] -
                   locs[((long long)b * NN + jc) * 2 + 0];
        float dy = locs[((long long)b * NN + ic) * 2 + 1] -
                   locs[((long long)b * NN + jc) * 2 + 1];
        distm[m] = sqrtf(dx * dx + dy * dy);
        slog[m]  = b2[0];
    }
    __syncthreads();

    // stage 2: gather emb_i|emb_j (f32 -> f16) into 64x512 A panel in LDS
#pragma unroll
    for (int it = 0; it < (MTILE * KDIM / 4) / 256; ++it) {   // 32 iters
        int v = tid + it * 256;          // 0..8191 vec4 slots
        int m = v >> 7;                  // edge row 0..63
        int k = (v & 127) << 2;          // 0..508 step 4
        int node = (k < 256) ? nodeI[m] : nodeJ[m];
        int koff = k & 255;
        f4v s = *(const f4v*)(emb + (((long long)b * NN + node) << 8) + koff);
        h4v d;
#pragma unroll
        for (int q = 0; q < 4; ++q) d[q] = (_Float16)s[q];
        *(h4v*)(&ldsA[m * LDS_STRIDE + k]) = d;
    }
    __syncthreads();

    // stage 3: (4x16 edges) x (64 cols per wave) GEMM, v_wmma_f32_16x16x32_f16
    const int lane = tid & 31;
    const int w    = tid >> 5;       // wave: hidden cols [w*64, w*64+64)
    const int g    = lane >> 4;      // half-wave group
    const int ln   = lane & 15;

    f8v acc[4][4];                   // [m-group][n-tile]
#pragma unroll
    for (int mg = 0; mg < 4; ++mg)
#pragma unroll
        for (int t = 0; t < 4; ++t) acc[mg][t] = (f8v)0.0f;

#pragma unroll 2
    for (int ks = 0; ks < 16; ++ks) {
        // A fragments (ISA 7.12.2, 16-bit A 16x32): V0-3 K=[8g..+7], V4-7 K=[8g+16..+23]
        const int akb = ks * 32 + g * 8;
        h16v A[4];
#pragma unroll
        for (int mg = 0; mg < 4; ++mg) {
            const _Float16* ap = &ldsA[(mg * 16 + ln) * LDS_STRIDE + akb];
            h8v alo = *(const h8v*)ap;
            h8v ahi = *(const h8v*)(ap + 16);
#pragma unroll
            for (int q = 0; q < 8; ++q) { A[mg][q] = alo[q]; A[mg][q + 8] = ahi[q]; }
        }
        // B fragments (32x16): V0-3 K=[8g..+7], V4-7 K=[8g+16..+23]; N=ln
        h16v Bf[4];
#pragma unroll
        for (int t = 0; t < 4; ++t) {
            const int n = w * 64 + t * 16 + ln;
            const _Float16* bp = W1t + (n << 9) + ks * 32 + g * 8;
            h8v blo = *(const h8v*)bp;
            h8v bhi = *(const h8v*)(bp + 16);
#pragma unroll
            for (int q = 0; q < 8; ++q) { Bf[t][q] = blo[q]; Bf[t][q + 8] = bhi[q]; }
        }
        // 16 WMMAs: each B fragment reused 4x, each A fragment reused 4x
#pragma unroll
        for (int t = 0; t < 4; ++t)
#pragma unroll
            for (int mg = 0; mg < 4; ++mg)
                acc[mg][t] = __builtin_amdgcn_wmma_f32_16x16x32_f16(
                    false, A[mg], false, Bf[t], (short)0, acc[mg][t], false, false);
    }

    // stage 4: + dist * W1[512,n] + b1, relu, * W2[n], reduce over n
    float p[4][8];
#pragma unroll
    for (int mg = 0; mg < 4; ++mg)
#pragma unroll
        for (int r = 0; r < 8; ++r) p[mg][r] = 0.0f;

    float dl[4][8];
#pragma unroll
    for (int mg = 0; mg < 4; ++mg)
#pragma unroll
        for (int r = 0; r < 8; ++r) dl[mg][r] = distm[mg * 16 + r + 8 * g];

#pragma unroll
    for (int t = 0; t < 4; ++t) {
        const int n = w * 64 + t * 16 + ln;
        float wl = W1[(long long)KDIM * HIDDEN + n];   // row k=512 (dist column)
        float bv = b1[n];
        float w2 = W2[n];
#pragma unroll
        for (int mg = 0; mg < 4; ++mg)
#pragma unroll
            for (int r = 0; r < 8; ++r) {
                float h = acc[mg][t][r] + dl[mg][r] * wl + bv;  // D: M=mg*16+r+8g, N=ln
                h = h > 0.0f ? h : 0.0f;
                p[mg][r] += h * w2;
            }
    }

    // xor-shuffle reduce across the 16 lanes of each half-wave (wave32)
#pragma unroll
    for (int mg = 0; mg < 4; ++mg)
#pragma unroll
        for (int r = 0; r < 8; ++r) {
            float v = p[mg][r];
            v += __shfl_xor(v, 1, 32);
            v += __shfl_xor(v, 2, 32);
            v += __shfl_xor(v, 4, 32);
            v += __shfl_xor(v, 8, 32);
            p[mg][r] = v;
        }
    if (ln == 0) {                    // lanes 0 and 16: m = mg*16 + r + 8g
#pragma unroll
        for (int mg = 0; mg < 4; ++mg)
#pragma unroll
            for (int r = 0; r < 8; ++r)
                atomicAdd(&slog[mg * 16 + r + 8 * g], p[mg][r]);
    }
    __syncthreads();

    // stage 5: mask + bias + store (guarded)
    if (tid < MTILE) {
        const int e = e0 + tid;
        if (e < EE) {
            float v = validm[tid] ? (slog[tid] + (float)dbias[0])
                                  : __int_as_float(0xff800000);  // -inf
            out[(long long)b * EE + e] = v;
        }
    }
}

extern "C" void kernel_launch(void* const* d_in, const int* in_sizes, int n_in,
                              void* d_out, int out_size, void* d_ws, size_t ws_size,
                              hipStream_t stream) {
    const float*     emb   = (const float*)d_in[0];
    const float*     locs  = (const float*)d_in[1];
    const long long* el    = (const long long*)d_in[2];
    const int*       dbias = (const int*)d_in[3];
    const float*     W1    = (const float*)d_in[4];
    const float*     b1    = (const float*)d_in[5];
    const float*     W2    = (const float*)d_in[6];
    const float*     b2    = (const float*)d_in[7];

    _Float16* W1t = (_Float16*)d_ws;   // 512*512*2 = 512 KB scratch

    w1_transpose_f16<<<(KDIM * HIDDEN + 255) / 256, 256, 0, stream>>>(W1, W1t);

    dim3 grid((EE + MTILE - 1) / MTILE, BB);   // 32 x 128 = 4096 blocks
    edge_mlp_wmma<<<grid, 256, 0, stream>>>(emb, locs, el, dbias,
                                            W1, b1, W2, b2, W1t,
                                            (float*)d_out);
}
